// MambaBlock_3745211482479
// MI455X (gfx1250) — matrix operand order repaired and checked by reading, not compiled, not run on hardware
//
#include <hip/hip_runtime.h>
#include <hip/hip_bf16.h>

// ---------------- problem constants (from reference) ----------------
#define D_MODEL   2048
#define D_INNER   4096
#define NXZ       (2 * D_INNER)   // 8192
#define D_STATE   16
#define D_CONV    4
#define DT_RANK   128
#define XDBL_PAD  256             // DT_RANK + 2*D_STATE = 160, padded to 256
#define BATCH     2
#define SEQLEN    4096
#define TOKENS    (BATCH * SEQLEN) // 8192
#define LN_EPS    1e-6f

typedef _Float16 half_t;
typedef __attribute__((ext_vector_type(16))) _Float16 v16h;
typedef __attribute__((ext_vector_type(8)))  _Float16 v8h;
typedef __attribute__((ext_vector_type(8)))  float    v8f;

// ======================================================================
// Generic fp32 -> f16 weight conversion with row zero-padding.
// ======================================================================
__global__ __launch_bounds__(256) void convert_pad_f16_kernel(
    const float* __restrict__ src, half_t* __restrict__ dst,
    long srcRows, long cols, long dstRows) {
  long i = (long)blockIdx.x * blockDim.x + threadIdx.x;
  long total = dstRows * cols;
  if (i >= total) return;
  long r = i / cols;
  dst[i] = (r < srcRows) ? (half_t)src[i] : (half_t)0.0f;
}

// ======================================================================
// LayerNorm over D_MODEL, fp32 in -> f16 out. One block per token.
// ======================================================================
__global__ __launch_bounds__(256) void layernorm_f16_kernel(
    const float* __restrict__ x, const float* __restrict__ w,
    const float* __restrict__ b, half_t* __restrict__ out) {
  __shared__ float red[256];
  const long t = blockIdx.x;
  const float* row = x + t * (long)D_MODEL;

  float s = 0.0f;
  for (int i = threadIdx.x; i < D_MODEL; i += 256) s += row[i];
  red[threadIdx.x] = s;
  __syncthreads();
  for (int st = 128; st > 0; st >>= 1) {
    if (threadIdx.x < st) red[threadIdx.x] += red[threadIdx.x + st];
    __syncthreads();
  }
  const float mu = red[0] * (1.0f / D_MODEL);
  __syncthreads();

  float v = 0.0f;
  for (int i = threadIdx.x; i < D_MODEL; i += 256) {
    float d = row[i] - mu;
    v += d * d;
  }
  red[threadIdx.x] = v;
  __syncthreads();
  for (int st = 128; st > 0; st >>= 1) {
    if (threadIdx.x < st) red[threadIdx.x] += red[threadIdx.x + st];
    __syncthreads();
  }
  const float rstd = rsqrtf(red[0] * (1.0f / D_MODEL) + LN_EPS);

  for (int i = threadIdx.x; i < D_MODEL; i += 256) {
    out[t * (long)D_MODEL + i] = (half_t)((row[i] - mu) * rstd * w[i] + b[i]);
  }
}

// ======================================================================
// WMMA GEMM with async global->LDS double buffering (ASYNCcnt-tracked):
//   C[M,N] = A[M,K] (f16 row-major) x W[N,K]^T (f16 row-major)
//   (+ optional fp32 residual fused into the store)
//
// 256 threads = 8 waves; wave tile 32(M) x 64(N); block tile 128x128.
// M, N multiples of 128; K multiple of 32.
//
// Per 32-K stage the block stages a 128x32 A tile and 128x32 B tile into
// LDS via global_load_async_to_lds_b128 (4 x 16B per thread), waits on
// ASYNCcnt, then feeds fragments from LDS. LDS rows padded to 40 halves
// so half-wave b128 fragment reads hit distinct stride-4 bank groups.
//
// Fragment layouts per CDNA5 ISA (CDNA5 ISA):
//  A (16x32 f16): lanes 0-15 = rows; g=0: K 0..7 / 16..23, g=1: K 8..15 / 24..31
//  B (32x16 f16): lane%16 = column; same K split as A (g*8..g*8+7 / 16+g*8..16+g*8+7)
//  C/D (16x16 f32): VGPR r -> row g*8+r, lane%16 = column.
// ======================================================================
#define KSTEP      32
#define LDS_STRIDE 40   // halves per LDS tile row (32 data + 8 pad)

__global__ __launch_bounds__(256) void gemm_wmma_f16_kernel(
    const half_t* __restrict__ A, const half_t* __restrict__ W,
    float* __restrict__ C, const float* __restrict__ residual,
    int M, int N, int K) {
  __shared__ __align__(128) half_t As[2][128 * LDS_STRIDE];
  __shared__ __align__(128) half_t Bs[2][128 * LDS_STRIDE];

  const int lane  = threadIdx.x & 31;
  const int wave  = threadIdx.x >> 5;
  const int waveM = wave & 3;   // 4 waves along M
  const int waveN = wave >> 2;  // 2 waves along N
  const int g     = lane >> 4;  // half-wave group
  const int ln    = lane & 15;

  const int mBase = blockIdx.y * 128 + waveM * 32;  // two 16-row tiles
  const int nBase = blockIdx.x * 128 + waveN * 64;  // four 16-col tiles

  const v8f vzero = {0.f, 0.f, 0.f, 0.f, 0.f, 0.f, 0.f, 0.f};
  v8f acc[2][4];
#pragma unroll
  for (int mt = 0; mt < 2; ++mt)
#pragma unroll
    for (int nt = 0; nt < 4; ++nt) acc[mt][nt] = vzero;

  const half_t* Ag = A + (long)blockIdx.y * 128 * K;  // block's 128 A rows
  const half_t* Wg = W + (long)blockIdx.x * 128 * K;  // block's 128 W rows

  // Issue one 128x32 tile (512 x 16B chunks, 2 chunks per thread) as
  // async global->LDS copies tracked by ASYNCcnt.
  auto issue_tile = [&](const half_t* gbase, half_t* lbase, int k0) {
#pragma unroll
    for (int j = 0; j < 2; ++j) {
      const int c   = (int)threadIdx.x + j * 256;  // chunk id 0..511
      const int row = c >> 2;                      // 4 chunks per row
      const int off = (c & 3) * 8;                 // halves within row
      const half_t* gp = gbase + (long)row * K + k0 + off;
      const unsigned lo =
          (unsigned)(size_t)(lbase + row * LDS_STRIDE + off);
      asm volatile("global_load_async_to_lds_b128 %0, %1, off"
                   :: "v"(lo), "v"(gp)
                   : "memory");
    }
  };

  issue_tile(Ag, &As[0][0], 0);
  issue_tile(Wg, &Bs[0][0], 0);

  const int nk = K / KSTEP;
  for (int ks = 0; ks < nk; ++ks) {
    const int buf = ks & 1;
    if (ks + 1 < nk) {
      issue_tile(Ag, &As[buf ^ 1][0], (ks + 1) * KSTEP);
      issue_tile(Wg, &Bs[buf ^ 1][0], (ks + 1) * KSTEP);
      // 8 async loads in flight; oldest 4 (current buffers) complete in order
      asm volatile("s_wait_asynccnt 0x4" ::: "memory");
    } else {
      asm volatile("s_wait_asynccnt 0x0" ::: "memory");
    }
    __syncthreads();

    const half_t* Abuf = &As[buf][0];
    const half_t* Bbuf = &Bs[buf][0];

    v16h a[2];
#pragma unroll
    for (int mt = 0; mt < 2; ++mt) {
      const half_t* p = Abuf + (waveM * 32 + mt * 16 + ln) * LDS_STRIDE + g * 8;
      v8h lo = *(const v8h*)p;
      v8h hi = *(const v8h*)(p + 16);
#pragma unroll
      for (int i = 0; i < 8; ++i) { a[mt][i] = lo[i]; a[mt][8 + i] = hi[i]; }
    }

#pragma unroll
    for (int nt = 0; nt < 4; ++nt) {
      const half_t* p = Bbuf + (waveN * 64 + nt * 16 + ln) * LDS_STRIDE + g * 8;
      v8h blo = *(const v8h*)p;
      v8h bhi = *(const v8h*)(p + 16);
      v16h bf;
#pragma unroll
      for (int i = 0; i < 8; ++i) { bf[i] = blo[i]; bf[8 + i] = bhi[i]; }
      acc[0][nt] = __builtin_amdgcn_wmma_f32_16x16x32_f16(
          false, a[0], false, bf, (short)0, acc[0][nt], false, false);
      acc[1][nt] = __builtin_amdgcn_wmma_f32_16x16x32_f16(
          false, a[1], false, bf, (short)0, acc[1][nt], false, false);
    }
    __syncthreads();  // all waves done reading buf before it is re-filled
  }

  // ---- epilogue: C rows g*8+r, column = lane%16 ----
#pragma unroll
  for (int mt = 0; mt < 2; ++mt) {
#pragma unroll
    for (int nt = 0; nt < 4; ++nt) {
      const int m0 = mBase + mt * 16 + g * 8;
      const int n  = nBase + nt * 16 + ln;
#pragma unroll
      for (int r = 0; r < 8; ++r) {
        long off = (long)(m0 + r) * N + n;
        float v = acc[mt][nt][r];
        if (residual) v += residual[off];
        C[off] = v;
      }
    }
  }
}

// ======================================================================
// Causal depthwise conv (D_CONV=4) + bias + SiLU.
// ======================================================================
__global__ __launch_bounds__(256) void conv_silu_kernel(
    const float* __restrict__ xz, const float* __restrict__ cw,
    const float* __restrict__ cb, float* __restrict__ xc,
    half_t* __restrict__ xc16) {
  long i = (long)blockIdx.x * blockDim.x + threadIdx.x;
  if (i >= (long)TOKENS * D_INNER) return;
  const int d = (int)(i % D_INNER);
  const long t = i / D_INNER;
  const int l = (int)(t % SEQLEN);

  float s = cb[d];
#pragma unroll
  for (int j = 0; j < D_CONV; ++j) {
    int off = j - (D_CONV - 1);
    if (l + off >= 0) s += xz[(t + off) * (long)NXZ + d] * cw[d * D_CONV + j];
  }
  float sv = s / (1.0f + __expf(-s));  // SiLU
  xc[i]   = sv;
  xc16[i] = (half_t)sv;
}

// ======================================================================
// Extract dt_in = x_dbl[:, :DT_RANK] as f16 for the dt GEMM.
// ======================================================================
__global__ __launch_bounds__(256) void dtin_f16_kernel(
    const float* __restrict__ xdbl, half_t* __restrict__ dtin) {
  long i = (long)blockIdx.x * blockDim.x + threadIdx.x;
  if (i >= (long)TOKENS * DT_RANK) return;
  long t = i / DT_RANK;
  int  r = (int)(i % DT_RANK);
  dtin[i] = (half_t)xdbl[t * (long)XDBL_PAD + r];
}

// ======================================================================
// dt = softplus(dt_raw + dt_proj_b), in place on [TOKENS, D_INNER].
// ======================================================================
__global__ __launch_bounds__(256) void softplus_kernel(
    float* __restrict__ dt, const float* __restrict__ bias) {
  long i = (long)blockIdx.x * blockDim.x + threadIdx.x;
  if (i >= (long)TOKENS * D_INNER) return;
  const int d = (int)(i % D_INNER);
  float x = dt[i] + bias[d];
  dt[i] = (x > 20.0f) ? x : log1pf(__expf(x));
}

// ======================================================================
// Selective scan: one thread per (batch, channel). state[16] in regs.
// ======================================================================
__global__ __launch_bounds__(256) void scan_kernel(
    const float* __restrict__ xc, const float* __restrict__ dt,
    const float* __restrict__ xdbl, const float* __restrict__ A_log,
    float* __restrict__ y) {
  int idx = blockIdx.x * blockDim.x + threadIdx.x;
  if (idx >= BATCH * D_INNER) return;
  const int d = idx % D_INNER;
  const int b = idx / D_INNER;

  float negA[D_STATE];
#pragma unroll
  for (int n = 0; n < D_STATE; ++n) negA[n] = -__expf(A_log[d * D_STATE + n]);
  float st[D_STATE];
#pragma unroll
  for (int n = 0; n < D_STATE; ++n) st[n] = 0.0f;

  for (int l = 0; l < SEQLEN; ++l) {
    const long t = (long)b * SEQLEN + l;
    const float dtv = dt[t * D_INNER + d];
    const float u   = xc[t * D_INNER + d];
    const float dtu = dtv * u;
    const float* Bp = xdbl + t * (long)XDBL_PAD + DT_RANK;
    const float* Cp = Bp + D_STATE;
    float accv = 0.0f;
#pragma unroll
    for (int n = 0; n < D_STATE; ++n) {
      st[n] = st[n] * __expf(dtv * negA[n]) + dtu * Bp[n];
      accv += st[n] * Cp[n];
    }
    y[t * D_INNER + d] = accv;
  }
}

// ======================================================================
// Gating: y = (y + D*xc) * silu(z);  z from xz[:, D_INNER:]. Write f16.
// ======================================================================
__global__ __launch_bounds__(256) void gate_kernel(
    const float* __restrict__ y, const float* __restrict__ xc,
    const float* __restrict__ Dp, const float* __restrict__ xz,
    half_t* __restrict__ y16) {
  long i = (long)blockIdx.x * blockDim.x + threadIdx.x;
  if (i >= (long)TOKENS * D_INNER) return;
  const int d = (int)(i % D_INNER);
  const long t = i / D_INNER;
  float z = xz[t * (long)NXZ + D_INNER + d];
  float yv = y[i] + Dp[d] * xc[i];
  y16[i] = (half_t)(yv * (z / (1.0f + __expf(-z))));
}

// ======================================================================
// Host-side orchestration
// ======================================================================
extern "C" void kernel_launch(void* const* d_in, const int* in_sizes, int n_in,
                              void* d_out, int out_size, void* d_ws, size_t ws_size,
                              hipStream_t stream) {
  (void)in_sizes; (void)n_in; (void)out_size; (void)ws_size;

  const float* hidden    = (const float*)d_in[0];   // [TOKENS, D_MODEL]
  const float* ln_w      = (const float*)d_in[1];
  const float* ln_b      = (const float*)d_in[2];
  const float* in_proj_w = (const float*)d_in[3];   // [NXZ, D_MODEL]
  const float* conv_w    = (const float*)d_in[4];   // [D_INNER, D_CONV]
  const float* conv_b    = (const float*)d_in[5];
  const float* x_proj_w  = (const float*)d_in[6];   // [160, D_INNER]
  const float* dt_proj_w = (const float*)d_in[7];   // [D_INNER, DT_RANK]
  const float* dt_proj_b = (const float*)d_in[8];
  const float* A_log     = (const float*)d_in[9];   // [D_INNER, D_STATE]
  const float* D_param   = (const float*)d_in[10];
  const float* out_proj_w= (const float*)d_in[11];  // [D_MODEL, D_INNER]
  float* out = (float*)d_out;                       // [TOKENS, D_MODEL]

  // ---- workspace carve-up (256B aligned) ----
  size_t off = 0;
  auto alloc = [&](size_t bytes) -> char* {
    char* p = (char*)d_ws + off;
    off += (bytes + 255) & ~(size_t)255;
    return p;
  };
  half_t* h16    = (half_t*)alloc((size_t)TOKENS * D_MODEL * 2);
  half_t* win16  = (half_t*)alloc((size_t)NXZ * D_MODEL * 2);
  half_t* wx16   = (half_t*)alloc((size_t)XDBL_PAD * D_INNER * 2);
  half_t* wdt16  = (half_t*)alloc((size_t)D_INNER * DT_RANK * 2);
  half_t* wout16 = (half_t*)alloc((size_t)D_MODEL * D_INNER * 2);
  float*  xz     = (float*)alloc((size_t)TOKENS * NXZ * 4);
  float*  xc     = (float*)alloc((size_t)TOKENS * D_INNER * 4);
  half_t* xc16   = (half_t*)alloc((size_t)TOKENS * D_INNER * 2);
  float*  xdbl   = (float*)alloc((size_t)TOKENS * XDBL_PAD * 4);
  half_t* dtin16 = (half_t*)alloc((size_t)TOKENS * DT_RANK * 2);
  float*  dtbuf  = (float*)alloc((size_t)TOKENS * D_INNER * 4);
  float*  ybuf   = (float*)alloc((size_t)TOKENS * D_INNER * 4);
  half_t* y16    = (half_t*)alloc((size_t)TOKENS * D_INNER * 2);

  const int TPB = 256;
  auto blocks = [](long total, int tpb) { return (int)((total + tpb - 1) / tpb); };

  // 0) weight conversion (f32 -> f16, pad x_proj rows 160 -> 256 with zeros)
  convert_pad_f16_kernel<<<blocks((long)NXZ * D_MODEL, TPB), TPB, 0, stream>>>(
      in_proj_w, win16, NXZ, D_MODEL, NXZ);
  convert_pad_f16_kernel<<<blocks((long)XDBL_PAD * D_INNER, TPB), TPB, 0, stream>>>(
      x_proj_w, wx16, DT_RANK + 2 * D_STATE, D_INNER, XDBL_PAD);
  convert_pad_f16_kernel<<<blocks((long)D_INNER * DT_RANK, TPB), TPB, 0, stream>>>(
      dt_proj_w, wdt16, D_INNER, DT_RANK, D_INNER);
  convert_pad_f16_kernel<<<blocks((long)D_MODEL * D_INNER, TPB), TPB, 0, stream>>>(
      out_proj_w, wout16, D_MODEL, D_INNER, D_MODEL);

  // 1) LayerNorm -> h16
  layernorm_f16_kernel<<<TOKENS, TPB, 0, stream>>>(hidden, ln_w, ln_b, h16);

  // 2) xz = h @ in_proj_w^T   (M=8192, N=8192, K=2048)
  gemm_wmma_f16_kernel<<<dim3(NXZ / 128, TOKENS / 128), TPB, 0, stream>>>(
      h16, win16, xz, nullptr, TOKENS, NXZ, D_MODEL);

  // 3) depthwise causal conv + SiLU -> xc (f32 + f16)
  conv_silu_kernel<<<blocks((long)TOKENS * D_INNER, TPB), TPB, 0, stream>>>(
      xz, conv_w, conv_b, xc, xc16);

  // 4) x_dbl = xc @ x_proj_w^T   (M=8192, N=256(pad), K=4096)
  gemm_wmma_f16_kernel<<<dim3(XDBL_PAD / 128, TOKENS / 128), TPB, 0, stream>>>(
      xc16, wx16, xdbl, nullptr, TOKENS, XDBL_PAD, D_INNER);

  // 5) dt_in -> f16
  dtin_f16_kernel<<<blocks((long)TOKENS * DT_RANK, TPB), TPB, 0, stream>>>(
      xdbl, dtin16);

  // 6) dt_raw = dt_in @ dt_proj_w^T   (M=8192, N=4096, K=128)
  gemm_wmma_f16_kernel<<<dim3(D_INNER / 128, TOKENS / 128), TPB, 0, stream>>>(
      dtin16, wdt16, dtbuf, nullptr, TOKENS, D_INNER, DT_RANK);

  // 7) dt = softplus(dt_raw + dt_proj_b)
  softplus_kernel<<<blocks((long)TOKENS * D_INNER, TPB), TPB, 0, stream>>>(
      dtbuf, dt_proj_b);

  // 8) selective scan -> y
  scan_kernel<<<blocks(BATCH * D_INNER, TPB), TPB, 0, stream>>>(
      xc, dtbuf, xdbl, A_log, ybuf);

  // 9) gating -> y16
  gate_kernel<<<blocks((long)TOKENS * D_INNER, TPB), TPB, 0, stream>>>(
      ybuf, xc, D_param, xz, y16);

  // 10) out = hidden + y @ out_proj_w^T   (M=8192, N=2048, K=4096)
  gemm_wmma_f16_kernel<<<dim3(D_MODEL / 128, TOKENS / 128), TPB, 0, stream>>>(
      y16, wout16, out, hidden, TOKENS, D_MODEL, D_INNER);
}
